// MultiHeadAttention_25598005084384
// MI455X (gfx1250) — matrix operand order repaired and checked by reading, not compiled, not run on hardware
//
#include <hip/hip_runtime.h>
#include <hip/hip_bf16.h>

// ---- problem constants (match reference) ----
#define BDIM 4
#define SEQ  1024
#define DM   1024
#define NH   16
#define DH   64
#define DSP  512
#define NEG_INF_V (-1e9f)

typedef __bf16 bf16;
typedef __attribute__((ext_vector_type(16))) __bf16 bf16x16;
typedef __attribute__((ext_vector_type(8)))  __bf16 bf16x8;
typedef __attribute__((ext_vector_type(4)))  __bf16 bf16x4;
typedef __attribute__((ext_vector_type(8)))  float  f32x8;

// ---- WMMA helper (CDNA5 wave32, bf16 in / f32 accumulate) ----
__device__ __forceinline__ f32x8 wmma_bf16(bf16x16 a, bf16x16 b, f32x8 c) {
  // 8 args: (neg_a, A, neg_b, B, c_mod, C, reuse_a, reuse_b)
  return __builtin_amdgcn_wmma_f32_16x16x32_bf16(false, a, false, b, (short)0, c,
                                                 false, false);
}

// A-fragment: 16x32 bf16, lane(l&15)=row M, kb=(l>>4)*16;
// elems 0..15 -> K=k0+kb..+15 contiguous ; source K-contiguous.
__device__ __forceinline__ bf16x16 frag_a(const bf16* base, int lda, int rowBase,
                                          int k0, int lane) {
  const int row = rowBase + (lane & 15);
  const int kb  = k0 + ((lane >> 4) << 4);
  bf16x8 lo = *(const bf16x8*)(base + (size_t)row * lda + kb);
  bf16x8 hi = *(const bf16x8*)(base + (size_t)row * lda + kb + 8);
  bf16x16 r;
#pragma unroll
  for (int i = 0; i < 8; ++i) { r[i] = lo[i]; r[i + 8] = hi[i]; }
  return r;
}

// B-fragment: 32x16 bf16, lane(l&15)=col N, K=k0+(l>>4)*16 .. +15 contiguous.
// Source stored [N][K] row-major -> 32 contiguous bytes per lane.
__device__ __forceinline__ bf16x16 frag_b(const bf16* base, int ldb, int colBase,
                                          int k0, int lane) {
  const int col = colBase + (lane & 15);
  const int kl  = k0 + ((lane >> 4) << 4);
  const bf16x8* p = (const bf16x8*)(base + (size_t)col * ldb + kl);
  bf16x8 lo = p[0];
  bf16x8 hi = p[1];
  bf16x16 r;
#pragma unroll
  for (int i = 0; i < 8; ++i) { r[i] = lo[i]; r[i + 8] = hi[i]; }
  return r;
}

// ---- weight packing: W[K,N] fp32 -> Wt[N][K] bf16 (32x32 LDS tile transpose) ----
__global__ __launch_bounds__(256) void pack_w(const float* __restrict__ W,
                                              bf16* __restrict__ Wt, int K, int N) {
  __shared__ float t[32][33];
  const int n0 = blockIdx.x * 32, k0 = blockIdx.y * 32;
  const int tx = threadIdx.x & 31, ty = threadIdx.x >> 5;  // 8 rows / pass
#pragma unroll
  for (int r = ty; r < 32; r += 8) t[r][tx] = W[(size_t)(k0 + r) * N + n0 + tx];
  __syncthreads();
#pragma unroll
  for (int r = ty; r < 32; r += 8)
    Wt[(size_t)(n0 + r) * K + k0 + tx] = (bf16)t[tx][r];
}

// ---- unified WMMA GEMM, double-buffered LDS (B pre-packed bf16 [N][K]) ----
// A_BF16 : A is bf16 row-major, else fp32 (converted during LDS fill)
// EPI    : 0 none | 1 +bias | 2 relu(+bias)
// OUT    : 0 f32 [M,N] | 1 bf16 [M,N] | 2 bf16 split-heads [B,H,S,DH]
//          3 bf16 transposed heads [B,H,DH,S]
template <bool A_BF16, int EPI, int OUT>
__global__ __launch_bounds__(128) void wmma_gemm(
    const void* __restrict__ Abase, int lda, const bf16* __restrict__ Bt, int ldb,
    const float* __restrict__ bias, void* __restrict__ Out, int M, int N, int K) {
  __shared__ bf16 lds_a[2][64 * 32];
  __shared__ bf16 lds_b[2][64 * 32];

  const int tid  = threadIdx.x;
  const int lane = tid & 31;
  const int wid  = tid >> 5;
  const int wm   = (wid >> 1) * 32;  // 2x2 wave grid over 64x64 block tile
  const int wn   = (wid & 1) * 32;
  const int mBase = blockIdx.y * 64;
  const int nBase = blockIdx.x * 64;

  // per-thread tile-fill coordinates: 16 contiguous elems / thread
  const int fi  = tid << 4;
  const int fr  = fi >> 5;   // row (A) / n (B) : 0..63
  const int fc  = fi & 31;   // k offset within tile: 0 or 16

  // prefetch registers
  bf16x8 ra0, ra1, rb0, rb1;
  float4 fa[4];

  auto load_tiles = [&](int k0) {
    if (A_BF16) {
      const bf16* g = (const bf16*)Abase + (size_t)(mBase + fr) * lda + k0 + fc;
      ra0 = *(const bf16x8*)g;
      ra1 = *(const bf16x8*)(g + 8);
    } else {
      const float* g = (const float*)Abase + (size_t)(mBase + fr) * lda + k0 + fc;
#pragma unroll
      for (int j = 0; j < 4; ++j) fa[j] = ((const float4*)g)[j];
    }
    const bf16* gb = Bt + (size_t)(nBase + fr) * ldb + k0 + fc;
    rb0 = *(const bf16x8*)gb;
    rb1 = *(const bf16x8*)(gb + 8);
  };
  auto store_tiles = [&](int buf) {
    bf16* da = lds_a[buf] + fr * 32 + fc;
    if (A_BF16) {
      *(bf16x8*)da = ra0;
      *(bf16x8*)(da + 8) = ra1;
    } else {
#pragma unroll
      for (int j = 0; j < 4; ++j) {
        bf16x4 p;
        p[0] = (bf16)fa[j].x; p[1] = (bf16)fa[j].y;
        p[2] = (bf16)fa[j].z; p[3] = (bf16)fa[j].w;
        *(bf16x4*)(da + 4 * j) = p;
      }
    }
    bf16* db = lds_b[buf] + fr * 32 + fc;
    *(bf16x8*)db = rb0;
    *(bf16x8*)(db + 8) = rb1;
  };

  f32x8 acc[2][2];
#pragma unroll
  for (int i = 0; i < 2; ++i)
#pragma unroll
    for (int j = 0; j < 2; ++j)
#pragma unroll
      for (int r = 0; r < 8; ++r) acc[i][j][r] = 0.f;

  load_tiles(0);
  store_tiles(0);
  int cur = 0;
  for (int k0 = 0; k0 < K; k0 += 32) {
    __syncthreads();                 // lds[cur] ready for reads
    const int nk = k0 + 32;
    if (nk < K) load_tiles(nk);      // overlap next-tile loads with WMMA

    bf16x16 a0 = frag_a(lds_a[cur], 32, wm, 0, lane);
    bf16x16 a1 = frag_a(lds_a[cur], 32, wm + 16, 0, lane);
    bf16x16 b0 = frag_b(lds_b[cur], 32, wn, 0, lane);
    bf16x16 b1 = frag_b(lds_b[cur], 32, wn + 16, 0, lane);
    acc[0][0] = wmma_bf16(a0, b0, acc[0][0]);
    acc[0][1] = wmma_bf16(a0, b1, acc[0][1]);
    acc[1][0] = wmma_bf16(a1, b0, acc[1][0]);
    acc[1][1] = wmma_bf16(a1, b1, acc[1][1]);

    if (nk < K) store_tiles(cur ^ 1);  // other buffer: no barrier needed here
    cur ^= 1;
  }

  // C layout: VGPR r, lane l -> M = r + 8*(l>>4), N = l&15
#pragma unroll
  for (int tm = 0; tm < 2; ++tm)
#pragma unroll
    for (int tn = 0; tn < 2; ++tn) {
      const int mB = mBase + wm + tm * 16 + ((lane >> 4) << 3);
      const int n  = nBase + wn + tn * 16 + (lane & 15);
#pragma unroll
      for (int r = 0; r < 8; ++r) {
        const int m = mB + r;
        float v = acc[tm][tn][r];
        if (EPI == 1 || EPI == 2) v += bias[n];
        if (EPI == 2) v = fmaxf(v, 0.f);
        if (OUT == 0) {
          ((float*)Out)[(size_t)m * N + n] = v;
        } else if (OUT == 1) {
          ((bf16*)Out)[(size_t)m * N + n] = (bf16)v;
        } else if (OUT == 2) {
          const int b = m / SEQ, s = m % SEQ, h = n / DH, d = n % DH;
          ((bf16*)Out)[(((size_t)(b * NH + h) * SEQ) + s) * DH + d] = (bf16)v;
        } else if (OUT == 3) {
          const int b = m / SEQ, s = m % SEQ, h = n / DH, d = n % DH;
          ((bf16*)Out)[(((size_t)(b * NH + h) * DH) + d) * SEQ + s] = (bf16)v;
        }
      }
    }
}

// ---- fused attention: logits + mask + softmax + attn-write + ctx ----
// One block = 16 waves = 512 threads handles one (b,h) x 32-query stripe.
// Logits stripe (32x1024) lives entirely in registers (8 C-tiles / wave;
// 512-thread blocks keep the per-wave VGPR budget high enough to avoid spills).
// Probabilities staged bf16 in a 64 KB LDS stripe (CDNA5: 320 KB/WGP),
// then ctx = P @ V is computed from LDS with K split 2-way across waves.
__global__ __launch_bounds__(512) void fused_attention(
    const bf16* __restrict__ qh, const bf16* __restrict__ kh,
    const bf16* __restrict__ vT, const float* __restrict__ mask,
    float* __restrict__ attn, bf16* __restrict__ ctx) {
  extern __shared__ char smem[];
  bf16*  q_s  = (bf16*)smem;                   // [32][64]    4 KB
  bf16*  p_s  = (bf16*)(smem + 4 * 1024);      // [32][1024]  64 KB
  float* red  = (float*)(smem + 68 * 1024);    // [2][8][16]  1 KB
  float* cred = (float*)(smem + 69 * 1024);    // [8][256]    8 KB

  const int tid = threadIdx.x, lane = tid & 31, w = tid >> 5;
  const int lh = lane >> 4, ln = lane & 15;
  const int m0 = blockIdx.x * 32;
  const int z  = blockIdx.y;
  const int b  = z / NH, h = z % NH;

  const bf16* qz = qh + (size_t)z * SEQ * DH + (size_t)m0 * DH;
  const bf16* kz = kh + (size_t)z * SEQ * DH;
  const bf16* vz = vT + (size_t)z * DH * SEQ;

  // stage 32x64 q stripe (4 bf16 per thread)
  {
    const int i = tid << 2;
    const int r = i >> 6, c = i & 63;
    *(bf16x4*)(q_s + r * 64 + c) = *(const bf16x4*)(qz + (size_t)r * DH + c);
  }
  __syncthreads();

  // ---- phase 1: logits. wave -> 16-row band (w&1) x 128 cols (w>>1) ----
  const int mtile = w & 1;
  const int ngrp  = w >> 1;
  const int band  = mtile * 16;

  f32x8 acc[8];
#pragma unroll
  for (int t = 0; t < 8; ++t)
#pragma unroll
    for (int r = 0; r < 8; ++r) acc[t][r] = 0.f;

  bf16x16 a0 = frag_a(q_s, 64, band, 0, lane);
  bf16x16 a1 = frag_a(q_s, 64, band, 32, lane);
#pragma unroll
  for (int t = 0; t < 8; ++t) {
    const int nc = ngrp * 128 + t * 16;
    bf16x16 b0 = frag_b(kz, DH, nc, 0, lane);  // K frags straight from global
    bf16x16 b1 = frag_b(kz, DH, nc, 32, lane);
    acc[t] = wmma_bf16(a0, b0, acc[t]);
    acc[t] = wmma_bf16(a1, b1, acc[t]);
  }

  // scale + additive mask (mask is [B,1,S,S], L2-resident across the 16 heads)
  const float* mrow = mask + (size_t)b * SEQ * SEQ;
#pragma unroll
  for (int t = 0; t < 8; ++t) {
    const int nc = ngrp * 128 + t * 16 + ln;
#pragma unroll
    for (int r = 0; r < 8; ++r) {
      const int m = m0 + band + r + (lh << 3);
      acc[t][r] = acc[t][r] * 0.125f + mrow[(size_t)m * SEQ + nc] * NEG_INF_V;
    }
  }

  // ---- phase 2: softmax over the stripe ----
  float rmax[8];
#pragma unroll
  for (int r = 0; r < 8; ++r) {
    rmax[r] = acc[0][r];
#pragma unroll
    for (int t = 1; t < 8; ++t) rmax[r] = fmaxf(rmax[r], acc[t][r]);
#pragma unroll
    for (int off = 1; off < 16; off <<= 1)
      rmax[r] = fmaxf(rmax[r], __shfl_xor(rmax[r], off, 32));
  }
  if (ln == 0) {
#pragma unroll
    for (int r = 0; r < 8; ++r)
      red[(mtile * 8 + ngrp) * 16 + (lh << 3) + r] = rmax[r];
  }
  __syncthreads();
  float gmax[8];
#pragma unroll
  for (int r = 0; r < 8; ++r) {
    float mx = -3.4e38f;
#pragma unroll
    for (int g = 0; g < 8; ++g)
      mx = fmaxf(mx, red[(mtile * 8 + g) * 16 + (lh << 3) + r]);
    gmax[r] = mx;
  }
  __syncthreads();

  float rsum[8];
#pragma unroll
  for (int r = 0; r < 8; ++r) rsum[r] = 0.f;
#pragma unroll
  for (int t = 0; t < 8; ++t)
#pragma unroll
    for (int r = 0; r < 8; ++r) {
      acc[t][r] = __expf(acc[t][r] - gmax[r]);
      rsum[r] += acc[t][r];
    }
#pragma unroll
  for (int r = 0; r < 8; ++r)
#pragma unroll
    for (int off = 1; off < 16; off <<= 1) rsum[r] += __shfl_xor(rsum[r], off, 32);
  if (ln == 0) {
#pragma unroll
    for (int r = 0; r < 8; ++r)
      red[(mtile * 8 + ngrp) * 16 + (lh << 3) + r] = rsum[r];
  }
  __syncthreads();
  float inv[8];
#pragma unroll
  for (int r = 0; r < 8; ++r) {
    float s = 0.f;
#pragma unroll
    for (int g = 0; g < 8; ++g) s += red[(mtile * 8 + g) * 16 + (lh << 3) + r];
    inv[r] = 1.f / s;
  }

  // normalize; write attn (the only HBM pass over it) + bf16 stripe to LDS
  float* attnz = attn + (size_t)z * SEQ * SEQ;
#pragma unroll
  for (int t = 0; t < 8; ++t) {
    const int nc = ngrp * 128 + t * 16 + ln;
#pragma unroll
    for (int r = 0; r < 8; ++r) {
      const int lm = band + r + (lh << 3);
      const float v = acc[t][r] * inv[r];
      attnz[(size_t)(m0 + lm) * SEQ + nc] = v;
      p_s[lm * 1024 + nc] = (bf16)v;
    }
  }
  __syncthreads();

  // ---- phase 3: ctx = P(LDS) @ V. 2m x 4n tiles x 2 K-halves across waves ----
  const int mt2 = w & 1, nt2 = (w >> 1) & 3, khalf = w >> 3;
  f32x8 c;
#pragma unroll
  for (int r = 0; r < 8; ++r) c[r] = 0.f;
  for (int k0 = khalf * 512; k0 < khalf * 512 + 512; k0 += 32) {
    bf16x16 a = frag_a(p_s, 1024, mt2 * 16, k0, lane);
    bf16x16 bb = frag_b(vz, SEQ, nt2 * 16, k0, lane);  // V frags from global
    c = wmma_bf16(a, bb, c);
  }
  const int tileid = w & 7;
  if (khalf == 1) {
#pragma unroll
    for (int r = 0; r < 8; ++r) cred[tileid * 256 + lane * 8 + r] = c[r];
  }
  __syncthreads();
  if (khalf == 0) {
#pragma unroll
    for (int r = 0; r < 8; ++r) c[r] += cred[tileid * 256 + lane * 8 + r];
#pragma unroll
    for (int r = 0; r < 8; ++r) {
      const int sq = m0 + mt2 * 16 + r + (lh << 3);
      const int d  = h * DH + nt2 * 16 + ln;
      ctx[((size_t)b * SEQ + sq) * DM + d] = (bf16)c[r];
    }
  }
}

extern "C" void kernel_launch(void* const* d_in, const int* in_sizes, int n_in,
                              void* d_out, int out_size, void* d_ws, size_t ws_size,
                              hipStream_t stream) {
  const float* q    = (const float*)d_in[0];
  const float* mdm  = (const float*)d_in[1];
  const float* mask = (const float*)d_in[2];
  const float* Wsp  = (const float*)d_in[3];
  const float* bsp  = (const float*)d_in[4];
  const float* Wq   = (const float*)d_in[5];
  const float* bq   = (const float*)d_in[6];
  const float* Wk   = (const float*)d_in[7];
  const float* bk   = (const float*)d_in[8];
  const float* Wv   = (const float*)d_in[9];
  const float* bv   = (const float*)d_in[10];
  const float* Wo   = (const float*)d_in[11];
  const float* bo   = (const float*)d_in[12];

  float* out  = (float*)d_out;                  // [B,S,DM]
  float* attn = out + (size_t)BDIM * SEQ * DM;  // [B,H,S,S]

  char* ws   = (char*)d_ws;
  bf16* sp   = (bf16*)(ws);                     // [B*S, DSP]    4 MiB
  bf16* qh   = (bf16*)(ws + (4ull << 20));      // [B,H,S,DH]    8 MiB
  bf16* kh   = (bf16*)(ws + (12ull << 20));     // [B,H,S,DH]    8 MiB
  bf16* vT   = (bf16*)(ws + (20ull << 20));     // [B,H,DH,S]    8 MiB
  bf16* ctx  = (bf16*)(ws + (28ull << 20));     // [B,S,DM]      8 MiB
  bf16* WspT = (bf16*)(ws + (36ull << 20));     // [512][1024]   1 MiB
  bf16* WqT  = (bf16*)(ws + (37ull << 20));     // [1024][1024]  2 MiB
  bf16* WkT  = (bf16*)(ws + (39ull << 20));     // [1024][512]   1 MiB
  bf16* WvT  = (bf16*)(ws + (40ull << 20));     // [1024][512]   1 MiB
  bf16* WoT  = (bf16*)(ws + (41ull << 20));     // [1024][1024]  2 MiB

  const dim3 blk(128);

  // 0) pack all weights to bf16 [N][K] once (removes per-block transpose cost)
  pack_w<<<dim3(DSP / 32, SEQ / 32), dim3(256), 0, stream>>>(Wsp, WspT, SEQ, DSP);
  pack_w<<<dim3(DM / 32, DM / 32), dim3(256), 0, stream>>>(Wq, WqT, DM, DM);
  pack_w<<<dim3(DM / 32, DSP / 32), dim3(256), 0, stream>>>(Wk, WkT, DSP, DM);
  pack_w<<<dim3(DM / 32, DSP / 32), dim3(256), 0, stream>>>(Wv, WvT, DSP, DM);
  pack_w<<<dim3(DM / 32, DM / 32), dim3(256), 0, stream>>>(Wo, WoT, DM, DM);

  // 1) sp = relu(mdm @ Wsp + bsp)            M=4096 N=512 K=1024
  wmma_gemm<false, 2, 1><<<dim3(DSP / 64, BDIM * SEQ / 64), blk, 0, stream>>>(
      mdm, SEQ, WspT, SEQ, bsp, sp, BDIM * SEQ, DSP, SEQ);

  // 2) qh = split_heads(q @ Wq + bq)         M=4096 N=1024 K=1024
  wmma_gemm<false, 1, 2><<<dim3(DM / 64, BDIM * SEQ / 64), blk, 0, stream>>>(
      q, DM, WqT, DM, bq, qh, BDIM * SEQ, DM, DM);

  // 3) kh = split_heads(sp @ Wk + bk)        M=4096 N=1024 K=512
  wmma_gemm<true, 1, 2><<<dim3(DM / 64, BDIM * SEQ / 64), blk, 0, stream>>>(
      sp, DSP, WkT, DSP, bk, kh, BDIM * SEQ, DM, DSP);

  // 4) vT = transpose_heads(sp @ Wv + bv)    M=4096 N=1024 K=512
  wmma_gemm<true, 1, 3><<<dim3(DM / 64, BDIM * SEQ / 64), blk, 0, stream>>>(
      sp, DSP, WvT, DSP, bv, vT, BDIM * SEQ, DM, DSP);

  // 5) fused attention: writes attn (fp32, once) and ctx (bf16 concat-heads)
  fused_attention<<<dim3(SEQ / 32, BDIM * NH), dim3(512), 77 * 1024, stream>>>(
      qh, kh, vT, mask, attn, ctx);

  // 6) out = ctx @ Wo + bo                   M=4096 N=1024 K=1024
  wmma_gemm<true, 1, 0><<<dim3(DM / 64, BDIM * SEQ / 64), blk, 0, stream>>>(
      ctx, DM, WoT, DM, bo, out, BDIM * SEQ, DM, DM);
}
